// Model_39676907885470
// MI455X (gfx1250) — matrix operand order repaired and checked by reading, not compiled, not run on hardware
//
#include <hip/hip_runtime.h>
#include <hip/hip_bf16.h>

// ---- CDNA5 wave32 WMMA attention (fp32 in/out, f16 WMMA, f32 accum) ----
// x1:[B,H,S,D] x2:[B,H,S,D] x3:[B,H,D,S] mask:[S,S] -> out:[B,H,D,S]
// out[d,k] = sum_q V[d,q] * softmax_row_q(QK^T*scale+mask)[q,k]
// Softmax runs in log2 domain: Q pre-scaled by scale*log2(e), mask folded via
// fmac with log2(e), stats stored in log2 domain -> raw v_exp_f32 per element.

typedef __attribute__((ext_vector_type(16))) _Float16 v16h;
typedef __attribute__((ext_vector_type(8)))  _Float16 v8h;
typedef __attribute__((ext_vector_type(2)))  __fp16   h2raw;
typedef __attribute__((ext_vector_type(8)))  float    v8f;
typedef __attribute__((ext_vector_type(4)))  float    v4f;

static constexpr int S_DIM = 2048;
static constexpr int D_DIM = 128;
static constexpr float LOG2E  = 1.4426950408889634f;
static constexpr float SC2    = 0.12754199460617296f; // (1/sqrt(128))*log2(e)

static __device__ __forceinline__ float fast_exp2(float x) {
#if __has_builtin(__builtin_amdgcn_exp2f)
    return __builtin_amdgcn_exp2f(x);      // single v_exp_f32
#else
    return exp2f(x);
#endif
}

static __device__ __forceinline__ v16h cat16(v8h lo, v8h hi) {
    v16h r;
#pragma unroll
    for (int i = 0; i < 8; ++i) { r[i] = lo[i]; r[i + 8] = hi[i]; }
    return r;
}

static __device__ __forceinline__ v8f wmma_f16(v16h a, v16h b, v8f c) {
    // v_wmma_f32_16x16x32_f16  (neg_a, A, neg_b, B, c_mod, C, reuse_a, reuse_b)
    return __builtin_amdgcn_wmma_f32_16x16x32_f16(false, a, false, b, (short)0, c,
                                                  false, false);
}

// 8 floats -> 8 halves via v_cvt_pk_rtz_f16_f32 (4 packed converts)
static __device__ __forceinline__ v8h pack8(v4f a, v4f b) {
    union { h2raw h[4]; v8h v; } u;
    u.h[0] = __builtin_amdgcn_cvt_pkrtz(a[0], a[1]);
    u.h[1] = __builtin_amdgcn_cvt_pkrtz(a[2], a[3]);
    u.h[2] = __builtin_amdgcn_cvt_pkrtz(b[0], b[1]);
    u.h[3] = __builtin_amdgcn_cvt_pkrtz(b[2], b[3]);
    return u.v;
}

// ---------------------------------------------------------------------------
// Kernel 1: per-row softmax stats (m2_q, 1/l_q) -> packed float2 workspace.
// grid = BH * (S/128), 256 threads. Wave w owns 16 complete query rows.
// ---------------------------------------------------------------------------
__global__ __launch_bounds__(256) void attn_stats(
    const float* __restrict__ x1, const float* __restrict__ x2,
    const float* __restrict__ mask, float2* __restrict__ wstat) {
    __shared__ __align__(16) _Float16 Qs[128][136];
    __shared__ __align__(16) _Float16 Ks[64][136];

    const int tid  = threadIdx.x;
    const int lane = tid & 31, wave = tid >> 5;
    const int ln = lane & 15, lg = lane >> 4;
    const int bh = blockIdx.x >> 4;
    const int q0 = (blockIdx.x & 15) << 7;

    { // Q tile 128x128 f32 -> f16 LDS, pre-scaled by scale*log2(e)
        const float* src = x1 + ((size_t)bh * S_DIM + q0) * D_DIM;
#pragma unroll
        for (int j = 0; j < 8; ++j) {
            unsigned c = tid + 256 * j;          // 2048 chunks of 8 floats
            unsigned row = c >> 4, col = (c & 15) << 3;
            v4f a = *(const v4f*)(src + row * (unsigned)D_DIM + col) * SC2;
            v4f b = *(const v4f*)(src + row * (unsigned)D_DIM + col + 4) * SC2;
            *(v8h*)&Qs[row][col] = pack8(a, b);
        }
    }

    float mrun[8], lrun[8];
#pragma unroll
    for (int r = 0; r < 8; ++r) { mrun[r] = -1e30f; lrun[r] = 0.0f; }

    const int qrow = 16 * wave + ln;            // A-matrix row (m = lane%16)

    for (int k0 = 0; k0 < S_DIM; k0 += 64) {
        __syncthreads();
        { // K tile 64x128 f32 -> f16 LDS
            const float* src = x2 + ((size_t)bh * S_DIM + k0) * D_DIM;
#pragma unroll
            for (int j = 0; j < 4; ++j) {
                unsigned c = tid + 256 * j;      // 1024 chunks of 8 floats
                unsigned row = c >> 4, col = (c & 15) << 3;
                v4f a = *(const v4f*)(src + row * (unsigned)D_DIM + col);
                v4f b = *(const v4f*)(src + row * (unsigned)D_DIM + col + 4);
                *(v8h*)&Ks[row][col] = pack8(a, b);
            }
        }
        if (k0 + 64 < S_DIM)  // gfx1250 global_prefetch_b8 for next tile
            __builtin_prefetch(x2 + ((size_t)bh * S_DIM + k0 + 64) * D_DIM + tid * 16, 0, 0);
        __syncthreads();

        // scores (log2 domain) for the whole 64-key strip, kept in registers
        v8f sv[4];
#pragma unroll
        for (int t = 0; t < 4; ++t) {
            v8f acc = {0.f, 0.f, 0.f, 0.f, 0.f, 0.f, 0.f, 0.f};
#pragma unroll
            for (int ds = 0; ds < 4; ++ds) {     // D=128 in 4 K-steps of 32
                v8h alo = *(const v8h*)&Qs[qrow][ds * 32 + 8 * lg];
                v8h ahi = *(const v8h*)&Qs[qrow][ds * 32 + 16 + 8 * lg];
                v8h blo = *(const v8h*)&Ks[16 * t + ln][ds * 32 + 8 * lg];
                v8h bhi = *(const v8h*)&Ks[16 * t + ln][ds * 32 + 16 + 8 * lg];
                acc = wmma_f16(cat16(alo, ahi), cat16(blo, bhi), acc);
            }
            const unsigned kcol = k0 + 16 * t + ln;
#pragma unroll
            for (int r = 0; r < 8; ++r) {
                const unsigned q = q0 + 16 * wave + r + 8 * lg;
                sv[t][r] = fmaf(mask[q * (unsigned)S_DIM + kcol], LOG2E, acc[r]);
            }
        }
        // one butterfly per row per 64 keys (VALU-combine across tiles first)
#pragma unroll
        for (int r = 0; r < 8; ++r) {
            float rm = fmaxf(fmaxf(sv[0][r], sv[1][r]),
                             fmaxf(sv[2][r], sv[3][r]));
            rm = fmaxf(rm, __shfl_xor(rm, 1, 32));
            rm = fmaxf(rm, __shfl_xor(rm, 2, 32));
            rm = fmaxf(rm, __shfl_xor(rm, 4, 32));
            rm = fmaxf(rm, __shfl_xor(rm, 8, 32));
            float mnew = fmaxf(mrun[r], rm);
            float ps = fast_exp2(sv[0][r] - mnew) + fast_exp2(sv[1][r] - mnew) +
                       fast_exp2(sv[2][r] - mnew) + fast_exp2(sv[3][r] - mnew);
            ps += __shfl_xor(ps, 1, 32);
            ps += __shfl_xor(ps, 2, 32);
            ps += __shfl_xor(ps, 4, 32);
            ps += __shfl_xor(ps, 8, 32);
            lrun[r] = lrun[r] * fast_exp2(mrun[r] - mnew) + ps;
            mrun[r] = mnew;
        }
    }

    if (ln == 0) {                               // lanes 0 and 16 write stats
#pragma unroll
        for (int r = 0; r < 8; ++r) {
            const unsigned q = q0 + 16 * wave + r + 8 * lg;
            wstat[(unsigned)bh * S_DIM + q] = make_float2(mrun[r], 1.0f / lrun[r]);
        }
    }
}

// ---------------------------------------------------------------------------
// Kernel 2: O[d,k] = sum_q V[d,q] * P[q,k].  grid = BH*(S/128), 256 threads.
// Block owns 128 output columns; loops q in steps of 32 (WMMA K dim).
// ---------------------------------------------------------------------------
__global__ __launch_bounds__(256) void attn_out(
    const float* __restrict__ x1, const float* __restrict__ x2,
    const float* __restrict__ x3, const float* __restrict__ mask,
    const float2* __restrict__ wstat, float* __restrict__ out) {
    __shared__ __align__(16) _Float16 Ks[128][136];  // key block (loop invariant)
    __shared__ __align__(16) _Float16 Qs[32][136];   // current q tile
    __shared__ __align__(16) _Float16 Pt[128][40];   // P^T : [k_local][q_local]

    const int tid  = threadIdx.x;
    const int lane = tid & 31, wave = tid >> 5;
    const int ln = lane & 15, lg = lane >> 4;
    const int bh = blockIdx.x >> 4;
    const int kbase = (blockIdx.x & 15) << 7;

    { // key block 128x128 f32 -> f16 LDS, loaded once
        const float* src = x2 + ((size_t)bh * S_DIM + kbase) * D_DIM;
#pragma unroll
        for (int j = 0; j < 8; ++j) {
            unsigned c = tid + 256 * j;          // 2048 chunks of 8 floats
            unsigned row = c >> 4, col = (c & 15) << 3;
            v4f a = *(const v4f*)(src + row * (unsigned)D_DIM + col);
            v4f b = *(const v4f*)(src + row * (unsigned)D_DIM + col + 4);
            *(v8h*)&Ks[row][col] = pack8(a, b);
        }
    }

    v8f oacc[8];
#pragma unroll
    for (int t = 0; t < 8; ++t)
#pragma unroll
        for (int i = 0; i < 8; ++i) oacc[t][i] = 0.0f;

    for (int q0 = 0; q0 < S_DIM; q0 += 32) {
        __syncthreads();
        { // q tile 32x128 f32 -> f16 LDS, pre-scaled by scale*log2(e)
            const float* src = x1 + ((size_t)bh * S_DIM + q0) * D_DIM;
#pragma unroll
            for (int j = 0; j < 2; ++j) {
                unsigned c = tid + 256 * j;      // 512 chunks of 8 floats
                unsigned row = c >> 4, col = (c & 15) << 3;
                v4f a = *(const v4f*)(src + row * (unsigned)D_DIM + col) * SC2;
                v4f b = *(const v4f*)(src + row * (unsigned)D_DIM + col + 4) * SC2;
                *(v8h*)&Qs[row][col] = pack8(a, b);
            }
        }
        __syncthreads();

        // ---- stage 1: scores + normalize -> Pt.  wave w owns k-sub w. ----
#pragma unroll
        for (int qs = 0; qs < 2; ++qs) {
            v8f acc = {0.f, 0.f, 0.f, 0.f, 0.f, 0.f, 0.f, 0.f};
#pragma unroll
            for (int ds = 0; ds < 4; ++ds) {
                v8h alo = *(const v8h*)&Qs[16 * qs + ln][ds * 32 + 8 * lg];
                v8h ahi = *(const v8h*)&Qs[16 * qs + ln][ds * 32 + 16 + 8 * lg];
                v8h blo = *(const v8h*)&Ks[16 * wave + ln][ds * 32 + 8 * lg];
                v8h bhi = *(const v8h*)&Ks[16 * wave + ln][ds * 32 + 16 + 8 * lg];
                acc = wmma_f16(cat16(alo, ahi), cat16(blo, bhi), acc);
            }
            const unsigned kg = kbase + 16 * wave + ln;
            float pe[8];
#pragma unroll
            for (int r = 0; r < 8; ++r) {
                const unsigned q = q0 + 16 * qs + r + 8 * lg;
                float s = fmaf(mask[q * (unsigned)S_DIM + kg], LOG2E, acc[r]);
                float2 st = wstat[(unsigned)bh * S_DIM + q];
                pe[r] = fast_exp2(s - st.x) * st.y;
            }
            union { h2raw h[4]; v8h v; } pv;     // 8 consecutive q_local cols
            pv.h[0] = __builtin_amdgcn_cvt_pkrtz(pe[0], pe[1]);
            pv.h[1] = __builtin_amdgcn_cvt_pkrtz(pe[2], pe[3]);
            pv.h[2] = __builtin_amdgcn_cvt_pkrtz(pe[4], pe[5]);
            pv.h[3] = __builtin_amdgcn_cvt_pkrtz(pe[6], pe[7]);
            *(v8h*)&Pt[16 * wave + ln][16 * qs + 8 * lg] = pv.v;  // one b128 store
        }
        __syncthreads();

        // ---- stage 2: O += V(16d x 32q) * P(32q x 16k), wave w owns d-sub w.
        const float* vsrc =
            x3 + ((size_t)(bh * D_DIM) + 16 * wave + ln) * S_DIM + q0;
        v4f va0 = *(const v4f*)(vsrc + 8 * lg);
        v4f va1 = *(const v4f*)(vsrc + 8 * lg + 4);
        v4f vb0 = *(const v4f*)(vsrc + 16 + 8 * lg);
        v4f vb1 = *(const v4f*)(vsrc + 20 + 8 * lg);
        v16h a = cat16(pack8(va0, va1), pack8(vb0, vb1));
#pragma unroll
        for (int t = 0; t < 8; ++t) {
            v8h blo = *(const v8h*)&Pt[16 * t + ln][8 * lg];
            v8h bhi = *(const v8h*)&Pt[16 * t + ln][16 + 8 * lg];
            oacc[t] = wmma_f16(a, cat16(blo, bhi), oacc[t]);
        }
    }

    // ---- write O[d, kbase:kbase+128] ----
    float* orow = out + (size_t)(bh * D_DIM) * S_DIM;
#pragma unroll
    for (int t = 0; t < 8; ++t)
#pragma unroll
        for (int r = 0; r < 8; ++r) {
            const unsigned d = 16 * wave + r + 8 * lg;
            orow[d * (unsigned)S_DIM + kbase + 16 * t + ln] = oacc[t][r];
        }
}

extern "C" void kernel_launch(void* const* d_in, const int* in_sizes, int n_in,
                              void* d_out, int out_size, void* d_ws,
                              size_t ws_size, hipStream_t stream) {
    (void)in_sizes; (void)n_in; (void)out_size; (void)ws_size;
    const float* x1   = (const float*)d_in[0];
    const float* x2   = (const float*)d_in[1];
    const float* x3   = (const float*)d_in[2];
    const float* mask = (const float*)d_in[3];
    float* out = (float*)d_out;
    float2* wstat = (float2*)d_ws;            // [32][2048] (log2 row max, 1/row sum)

    const int nblocks = 32 * (S_DIM / 128);   // 512
    attn_stats<<<nblocks, 256, 0, stream>>>(x1, x2, mask, wstat);
    attn_out<<<nblocks, 256, 0, stream>>>(x1, x2, x3, mask, wstat, out);
}
